// GDULayer_11175504904824
// MI455X (gfx1250) — matrix operand order repaired and checked by reading, not compiled, not run on hardware
//
#include <hip/hip_runtime.h>
#include <hip/hip_bf16.h>
#include <stdint.h>

// ---- types for CDNA5 WMMA -------------------------------------------------
typedef __attribute__((ext_vector_type(16))) __bf16      v16bf;
typedef __attribute__((ext_vector_type(8)))  float       v8f;
typedef __attribute__((ext_vector_type(4)))  unsigned int v4u;

union Frag { v16bf v; v4u q[2]; };

// ---- problem constants ----------------------------------------------------
#define B_    65536
#define F_    512
#define G_    8
#define D_    16
#define O_    32
#define NCOMB 384          // 128 (xv) + 256 (y) combined N
#define NT    24           // N tiles of 16
#define KC    16           // K chunks of 32 (F = 512)
#define MROWS 32           // rows of x per workgroup (2 WMMA row-tiles)
#define XPITCH 520         // 512 + 8 bf16 pad (row stride stays 16B-aligned)

// inv_2s2 = 1/(2*0.5^2) = 2.0
#define INV2S2 2.0f

__device__ __forceinline__ unsigned short f2bf(float f) {
  union { float f; unsigned int u; } cv; cv.f = f;
  unsigned int u = cv.u;
  return (unsigned short)((u + 0x7FFFu + ((u >> 16) & 1u)) >> 16); // RNE
}

// ---------------------------------------------------------------------------
// Kernel 1: pack V[G,F,D] and W[G,O,F] (fp32) into bf16 B-fragments laid out
// in contiguous-k per-lane order (same k order as the A loads in gdu_main):
//   fragment (nt, kc): 32 lanes x 16 bf16.  lane l -> col n = nt*16+(l&15),
//   element e -> K = kc*32 + (l>>4)*16 + e.
// Combined column space: n<128 -> V (g=n>>4, d=n&15), else W (g,o).
// ---------------------------------------------------------------------------
__global__ __launch_bounds__(256) void gdu_pack(
    const float* __restrict__ V, const float* __restrict__ W,
    unsigned short* __restrict__ Bpack)
{
  int tid = blockIdx.x * blockDim.x + threadIdx.x;   // 0 .. 196607
  if (tid >= NT * KC * 32 * 16) return;
  int e    = tid & 15;
  int lane = (tid >> 4) & 31;
  int frag = tid >> 9;            // nt*16 + kc
  int kc   = frag & 15;
  int nt   = frag >> 4;
  int n = nt * 16 + (lane & 15);
  int k = kc * 32 + (lane >> 4) * 16 + e;
  float src;
  if (n < 128) {                      // V part: V[g, k, d]
    int g = n >> 4, d = n & 15;
    src = V[(g * F_ + k) * D_ + d];
  } else {                            // W part: W[g, o, k]
    int m = n - 128;
    int g = m >> 5, o = m & 31;
    src = W[(g * O_ + o) * F_ + k];
  }
  Bpack[tid] = f2bf(src);
}

// ---------------------------------------------------------------------------
// Kernel 2: v2[g,d] = sum_f V^2 ; kVV[g] = mean_{d,e} exp(-d2(Vd,Ve)*2)
// One block of 128 threads, deterministic LDS reduction (no FP atomics).
// ---------------------------------------------------------------------------
__global__ __launch_bounds__(128) void gdu_prep(
    const float* __restrict__ V, float* __restrict__ v2out,
    float* __restrict__ kVVout)
{
  __shared__ float v2s[128];
  __shared__ float part[128];
  int t = threadIdx.x;
  int g = t >> 4, d = t & 15;
  const float* Vg = V + (size_t)g * F_ * D_;
  float s = 0.f;
  for (int f = 0; f < F_; ++f) { float v = Vg[f * D_ + d]; s += v * v; }
  v2s[t] = s;
  v2out[t] = s;
  __syncthreads();
  float acc = 0.f;
  for (int e = 0; e < D_; ++e) {
    float dot = 0.f;
    for (int f = 0; f < F_; ++f) dot += Vg[f * D_ + d] * Vg[f * D_ + e];
    float d2 = v2s[g * 16 + d] + v2s[g * 16 + e] - 2.f * dot;
    acc += __expf(-INV2S2 * d2);
  }
  part[t] = acc;
  __syncthreads();
  if (t < G_) {
    float m = 0.f;
    for (int i = 0; i < 16; ++i) m += part[t * 16 + i];
    kVVout[t] = m * (1.0f / 256.0f);
  }
}

// ---------------------------------------------------------------------------
// Kernel 3: fused main. Block = 32 rows of x (two 16-row WMMA tiles).
// 8 waves x 3 N-tiles; each B fragment feeds 2 WMMAs (row-tiles 0 and 1).
// bf16 WMMA with fp32 accumulate, then RBF/softmax/gating epilogue.
// ---------------------------------------------------------------------------
__global__ __launch_bounds__(256) void gdu_main(
    const float* __restrict__ x, const unsigned short* __restrict__ Bpack,
    const float* __restrict__ v2, const float* __restrict__ kVV,
    const float* __restrict__ bias, float* __restrict__ out)
{
  __shared__ unsigned short xs[MROWS][XPITCH];  // bf16 x tile   (33.3 KB)
  __shared__ float cstage[MROWS][NCOMB];        // fp32 staging  (49.2 KB)
  __shared__ float xpart[MROWS][8];
  __shared__ float x2s[MROWS];
  __shared__ float betaRaw[MROWS][8];
  __shared__ float betaS[MROWS][8];

  const int tid = threadIdx.x;
  const long row0 = (long)blockIdx.x * MROWS;

  // ---- stage x tile -> bf16 LDS, x^2 partials (8 threads x 64 elems/row) ----
  {
    const int row = tid >> 3;        // 0..31
    const int seg = tid & 7;         // 64 elems each
    const float* xr = x + (row0 + row) * F_ + seg * 64;
    float s = 0.f;
    #pragma unroll
    for (int i = 0; i < 64; i += 4) {
      float4 v4 = *(const float4*)(xr + i);
      s += v4.x * v4.x + v4.y * v4.y + v4.z * v4.z + v4.w * v4.w;
      int c0 = seg * 64 + i;
      xs[row][c0 + 0] = f2bf(v4.x);
      xs[row][c0 + 1] = f2bf(v4.y);
      xs[row][c0 + 2] = f2bf(v4.z);
      xs[row][c0 + 3] = f2bf(v4.w);
    }
    xpart[row][seg] = s;
  }
  __syncthreads();
  if (tid < MROWS) {
    float s = 0.f;
    #pragma unroll
    for (int i = 0; i < 8; ++i) s += xpart[tid][i];
    x2s[tid] = s;
  }

  // ---- WMMA main loop: wave owns 3 N-tiles x 2 row-tiles ----
  const int lane  = tid & 31;
  const int wave  = tid >> 5;
  const int nt0   = wave * 3;
  const int rowA  = lane & 15;
  const int khalf = lane >> 4;
  v8f a00 = 0.f, a01 = 0.f, a02 = 0.f;   // row-tile 0, nt 0..2
  v8f a10 = 0.f, a11 = 0.f, a12 = 0.f;   // row-tile 1, nt 0..2
  const unsigned short* arow0 = &xs[rowA][khalf * 16];
  const unsigned short* arow1 = &xs[16 + rowA][khalf * 16];

  #pragma unroll 2
  for (int kc = 0; kc < KC; ++kc) {
    // A fragments (contiguous k = 16*khalf + e per lane, same order as Bpack)
    Frag fa0, fa1;
    const unsigned short* ap0 = arow0 + kc * 32;
    const unsigned short* ap1 = arow1 + kc * 32;
    fa0.q[0] = *(const v4u*)(ap0);  fa0.q[1] = *(const v4u*)(ap0 + 8);
    fa1.q[0] = *(const v4u*)(ap1);  fa1.q[1] = *(const v4u*)(ap1 + 8);
    // B fragments: contiguous 32B per lane in packed workspace (L2-resident)
    const unsigned short* bp =
        Bpack + ((size_t)(nt0 * 16 + kc) * 512) + lane * 16;
    Frag b0, b1, b2;
    b0.q[0] = *(const v4u*)(bp);              b0.q[1] = *(const v4u*)(bp + 8);
    b1.q[0] = *(const v4u*)(bp + 16 * 512);   b1.q[1] = *(const v4u*)(bp + 16 * 512 + 8);
    b2.q[0] = *(const v4u*)(bp + 32 * 512);   b2.q[1] = *(const v4u*)(bp + 32 * 512 + 8);

    a00 = __builtin_amdgcn_wmma_f32_16x16x32_bf16(false, fa0.v, false, b0.v,
                                                  (short)0, a00, false, false);
    a10 = __builtin_amdgcn_wmma_f32_16x16x32_bf16(false, fa1.v, false, b0.v,
                                                  (short)0, a10, false, false);
    a01 = __builtin_amdgcn_wmma_f32_16x16x32_bf16(false, fa0.v, false, b1.v,
                                                  (short)0, a01, false, false);
    a11 = __builtin_amdgcn_wmma_f32_16x16x32_bf16(false, fa1.v, false, b1.v,
                                                  (short)0, a11, false, false);
    a02 = __builtin_amdgcn_wmma_f32_16x16x32_bf16(false, fa0.v, false, b2.v,
                                                  (short)0, a02, false, false);
    a12 = __builtin_amdgcn_wmma_f32_16x16x32_bf16(false, fa1.v, false, b2.v,
                                                  (short)0, a12, false, false);
  }

  // ---- stage accumulators (C/D layout: vgpr j -> row j + (lane>>4)*8, col lane&15) ----
  {
    const int col = lane & 15;
    const int rb  = khalf * 8;
    #pragma unroll
    for (int j = 0; j < 8; ++j) {
      cstage[rb + j][(nt0 + 0) * 16 + col] = a00[j];
      cstage[rb + j][(nt0 + 1) * 16 + col] = a01[j];
      cstage[rb + j][(nt0 + 2) * 16 + col] = a02[j];
      cstage[16 + rb + j][(nt0 + 0) * 16 + col] = a10[j];
      cstage[16 + rb + j][(nt0 + 1) * 16 + col] = a11[j];
      cstage[16 + rb + j][(nt0 + 2) * 16 + col] = a12[j];
    }
  }
  __syncthreads();

  // ---- phase 1: raw MMD score per (row, expert): 32x8 = 256 tasks ----
  {
    int r = tid >> 3, g = tid & 7;
    float x2 = x2s[r];
    float s = 0.f;
    #pragma unroll
    for (int d = 0; d < D_; ++d) {
      float xv = cstage[r][g * 16 + d];
      float d2 = x2 - 2.f * xv + v2[g * 16 + d];
      s += __expf(-INV2S2 * d2);
    }
    // beta = k(x,x) - 2*mean_d k(x,V) + kVV = 1 - (2/16)*s + kVV
    betaRaw[r][g] = 1.f + kVV[g] - s * 0.125f;
  }
  __syncthreads();

  // ---- phase 2: softmax over 8 experts per row ----
  if (tid < MROWS) {
    float m = betaRaw[tid][0];
    #pragma unroll
    for (int g = 1; g < G_; ++g) m = fmaxf(m, betaRaw[tid][g]);
    float e[G_]; float den = 0.f;
    #pragma unroll
    for (int g = 0; g < G_; ++g) { e[g] = __expf(betaRaw[tid][g] - m); den += e[g]; }
    float inv = 1.f / den;
    #pragma unroll
    for (int g = 0; g < G_; ++g) betaS[tid][g] = e[g] * inv;
  }
  __syncthreads();

  // ---- phase 3: gated mixture -> out[B, 32] (32x32 = 1024, 4 per thread) ----
  {
    int r = tid >> 3, o0 = tid & 7;
    #pragma unroll
    for (int h = 0; h < 4; ++h) {
      int o = o0 + h * 8;
      float p = 0.f;
      #pragma unroll
      for (int g = 0; g < G_; ++g)
        p += betaS[r][g] * (cstage[r][128 + g * 32 + o] + bias[g * 32 + o]);
      out[(row0 + r) * O_ + o] = p;
    }
  }
}

// ---------------------------------------------------------------------------
extern "C" void kernel_launch(void* const* d_in, const int* in_sizes, int n_in,
                              void* d_out, int out_size, void* d_ws, size_t ws_size,
                              hipStream_t stream) {
  (void)in_sizes; (void)n_in; (void)out_size; (void)ws_size;
  const float* x = (const float*)d_in[0];   // [B, F]
  const float* V = (const float*)d_in[1];   // [G, F, D]
  const float* W = (const float*)d_in[2];   // [G, O, F]
  const float* b = (const float*)d_in[3];   // [G, O]
  float* out = (float*)d_out;               // [B, O, 1] flat

  // workspace: Bpack (384 KB bf16) | v2 (128 f32) | kVV (8 f32)
  unsigned short* Bpack = (unsigned short*)d_ws;
  float* v2  = (float*)((char*)d_ws + (size_t)NT * KC * 32 * 16 * sizeof(unsigned short));
  float* kVV = v2 + 128;

  gdu_pack<<<768, 256, 0, stream>>>(V, W, Bpack);          // 768*256 = 196608 elems
  gdu_prep<<<1, 128, 0, stream>>>(V, v2, kVV);
  gdu_main<<<B_ / MROWS, 256, 0, stream>>>(x, Bpack, v2, kVV, b, out);
}
